// BlocksCore_13812614824607
// MI455X (gfx1250) — matrix operand order repaired and checked by reading, not compiled, not run on hardware
//
#include <hip/hip_runtime.h>
#include <hip/hip_bf16.h>

typedef __attribute__((ext_vector_type(16))) _Float16 v16h;
typedef __attribute__((ext_vector_type(8)))  _Float16 v8h;
typedef __attribute__((ext_vector_type(8)))  float    v8f;
typedef __attribute__((ext_vector_type(4)))  unsigned int u32x4;
typedef __attribute__((ext_vector_type(8)))  int          i32x8;
typedef __attribute__((ext_vector_type(4)))  int          i32x4;

static constexpr int    Bsz  = 4096;
static constexpr int    NINP = 512;
static constexpr int    NHID = 2048;
static constexpr int    NB   = 8;
static constexpr int    BS   = 256;
static constexpr int    AOUT = 1024;
static constexpr int    DKI  = 64;
static constexpr size_t HTOT = (size_t)Bsz * NHID;   // 8388608

__device__ __forceinline__ float sigm(float x) { return 1.0f / (1.0f + __expf(-x)); }

#define WMMA16(a, b, c) \
  __builtin_amdgcn_wmma_f32_16x16x32_f16(false, (a), false, (b), (short)0, (c), false, false)

// ---- CDNA5 async global->LDS copy (ASYNCcnt path, ISA §15.18.3 op 98) ----
__device__ __forceinline__ void async_ld_b128(void* lds, const void* gptr) {
  asm volatile("global_load_async_to_lds_b128 %0, %1, off"
               :: "v"((unsigned)(uintptr_t)lds), "v"(gptr)
               : "memory");
}
__device__ __forceinline__ void wait_async0() {
  asm volatile("s_wait_asynccnt 0x0" ::: "memory");
}

// ==================== weight packing into WMMA B fragment-record order ====================
// Record order: for matrix m: i = (((m*KT + kt)*NS + ns)*32 + lane)*16 + e
//   row K = kt*32 + (e>>3)*16 + (lane>>4)*8 + (e&7) ;  col N = ns*16 + (lane&15)
__device__ __forceinline__ void frag_decode(long i, int NS, int KT, int& m, int& row, int& col) {
  int  e  = (int)(i & 15);
  int  l  = (int)((i >> 4) & 31);
  long rc = i >> 9;
  int  ns = (int)(rc % NS);
  long t  = rc / NS;
  int  kt = (int)(t % KT);
  m   = (int)(t / KT);
  row = kt * 32 + ((e >> 3) << 4) + ((l >> 4) << 3) + (e & 7);
  col = ns * 16 + (l & 15);
}

// [Wk_i[1] | Wv_i[1]] : K=512, N=1088
__global__ void pack_wkv1_kernel(_Float16* dst, const float* Wk, const float* Wv) {
  const long n = 16L * 68 * 512;
  for (long i = (long)blockIdx.x * blockDim.x + threadIdx.x; i < n;
       i += (long)gridDim.x * blockDim.x) {
    int m, d, c;
    frag_decode(i, 68, 16, m, d, c);
    float v = (c < DKI) ? Wk[(size_t)NINP * DKI + (size_t)d * DKI + c]
                        : Wv[(size_t)NINP * AOUT + (size_t)d * AOUT + (c - DKI)];
    dst[i] = (_Float16)v;
  }
}

// Wq_i[k] : per-k K=256, N=64
__global__ void pack_wqi_kernel(_Float16* dst, const float* Wq) {
  const long n = (long)NB * 8 * 4 * 512;
  for (long i = (long)blockIdx.x * blockDim.x + threadIdx.x; i < n;
       i += (long)gridDim.x * blockDim.x) {
    int k, d, c;
    frag_decode(i, 4, 8, k, d, c);
    dst[i] = (_Float16)Wq[((size_t)k * BS + d) * DKI + c];
  }
}

// Wih[k]^T with gate-column permute: c -> g = (c&127)>>5 * 256 + (c>>7)*32 + (c&31)
__global__ void pack_wih_kernel(_Float16* dst, const float* W) {
  const long n = (long)NB * 32 * 64 * 512;
  for (long i = (long)blockIdx.x * blockDim.x + threadIdx.x; i < n;
       i += (long)gridDim.x * blockDim.x) {
    int k, d, c;
    frag_decode(i, 64, 32, k, d, c);
    int jt = c >> 7, r = c & 127;
    int g  = (r >> 5) * BS + jt * 32 + (r & 31);
    dst[i] = (_Float16)W[(((size_t)k << 10) + g) * 1024 + d];
  }
}

// Whh[k]^T with same gate-column permute: K=256
__global__ void pack_whh_kernel(_Float16* dst, const float* W) {
  const long n = (long)NB * 8 * 64 * 512;
  for (long i = (long)blockIdx.x * blockDim.x + threadIdx.x; i < n;
       i += (long)gridDim.x * blockDim.x) {
    int k, d, c;
    frag_decode(i, 64, 8, k, d, c);
    int jt = c >> 7, r = c & 127;
    int g  = (r >> 5) * BS + jt * 32 + (r & 31);
    dst[i] = (_Float16)W[(((size_t)k << 10) + g) * 256 + d];
  }
}

// [Wq_m|Wk_m|Wv_m][k] : per-k K=256, N=192
__global__ void pack_wqkvm_kernel(_Float16* dst, const float* Wq, const float* Wk,
                                  const float* Wv) {
  const long n = (long)NB * 8 * 12 * 512;
  for (long i = (long)blockIdx.x * blockDim.x + threadIdx.x; i < n;
       i += (long)gridDim.x * blockDim.x) {
    int k, d, c;
    frag_decode(i, 12, 8, k, d, c);
    size_t base = ((size_t)k * BS + d) * DKI;
    float v = (c < 64) ? Wq[base + c] : (c < 128) ? Wk[base + c - 64] : Wv[base + c - 128];
    dst[i] = (_Float16)v;
  }
}

// gate/fc paired columns: c -> j=c>>1, p=c&1 : K=64, N=512
__global__ void pack_wgf_kernel(_Float16* dst, const float* gate_w, const float* fc_w) {
  const long n = 2L * 32 * 512;
  for (long i = (long)blockIdx.x * blockDim.x + threadIdx.x; i < n;
       i += (long)gridDim.x * blockDim.x) {
    int m, d, c;
    frag_decode(i, 32, 2, m, d, c);
    int j = c >> 1;
    dst[i] = (_Float16)((c & 1) ? fc_w[j * 64 + d] : gate_w[j * 64 + d]);
  }
}

__global__ void cvt_f16_kernel(_Float16* dst, const float* src, long n) {
  for (long i = (long)blockIdx.x * blockDim.x + threadIdx.x; i < n;
       i += (long)gridDim.x * blockDim.x)
    dst[i] = (_Float16)src[i];
}

// ==================== generic f16 WMMA GEMM (async + double buffered) ====================
// C[m, cOffZ*z + n] = sum_k A[m, aOffZ*z + k] * Bfrag[z][k, n]
// tile 128(M) x 64(N); 8 waves 4x2; wave = 32x32 = 2x2 fragments.
__global__ __launch_bounds__(256) void gemm_f16_nt(const _Float16* __restrict__ A, int lda,
                                                   int aOffZ, const _Float16* __restrict__ Bf,
                                                   int NS, long bOffZ, float* __restrict__ C,
                                                   int ldc, int cOffZ, int K) {
  __shared__ _Float16 aL[2][128 * 32];
  __shared__ _Float16 bL[2][4 * 32 * 16];
  const int tid = threadIdx.x, lane = tid & 31, wave = tid >> 5;
  const int wm = wave >> 1, wn = wave & 1;
  const int m0 = blockIdx.x * 128, n0 = blockIdx.y * 64, ns0 = blockIdx.y * 4;
  const int z  = blockIdx.z;

  const int arow  = (tid >> 5) * 16 + (lane & 15);
  const int ahalf = lane >> 4;
  const _Float16* Ab = A + (size_t)(m0 + arow) * lda + (size_t)aOffZ * z + ahalf * 8;
  const _Float16* Bb = Bf + (size_t)bOffZ * z + (size_t)ns0 * 512 + tid * 8;
  const int nk = K >> 5;

  v8f acc[2][2] = {};
  // prologue: stage tile 0
  async_ld_b128(&aL[0][tid * 16], Ab);
  async_ld_b128(&aL[0][tid * 16 + 8], Ab + 16);
  async_ld_b128(&bL[0][tid * 8], Bb);
  wait_async0();
  __syncthreads();
  for (int kt = 0; kt < nk; kt++) {
    const int buf = kt & 1;
    if (kt + 1 < nk) {  // prefetch next tile into other buffer
      async_ld_b128(&aL[buf ^ 1][tid * 16], Ab + (kt + 1) * 32);
      async_ld_b128(&aL[buf ^ 1][tid * 16 + 8], Ab + (kt + 1) * 32 + 16);
      async_ld_b128(&bL[buf ^ 1][tid * 8], Bb + (size_t)(kt + 1) * NS * 512);
    }
    v16h a0 = *(const v16h*)&aL[buf][((wm * 2 + 0) * 32 + lane) * 16];
    v16h a1 = *(const v16h*)&aL[buf][((wm * 2 + 1) * 32 + lane) * 16];
    v16h b0 = *(const v16h*)&bL[buf][((wn * 2 + 0) * 32 + lane) * 16];
    v16h b1 = *(const v16h*)&bL[buf][((wn * 2 + 1) * 32 + lane) * 16];
    acc[0][0] = WMMA16(a0, b0, acc[0][0]);
    acc[1][0] = WMMA16(a1, b0, acc[1][0]);
    acc[0][1] = WMMA16(a0, b1, acc[0][1]);
    acc[1][1] = WMMA16(a1, b1, acc[1][1]);
    wait_async0();
    __syncthreads();
  }
  const int rbase = m0 + wm * 32 + (lane >> 4) * 8;
  const int cbase = cOffZ * z + n0 + wn * 32 + (lane & 15);
#pragma unroll
  for (int fm = 0; fm < 2; fm++)
#pragma unroll
    for (int fn = 0; fn < 2; fn++)
#pragma unroll
      for (int e = 0; e < 8; e++)
        C[(size_t)(rbase + fm * 16 + e) * ldc + cbase + fn * 16] = acc[fm][fn][e];
}

// ==================== input attention scale + top-k block mask ====================
__global__ void iatt_topk_kernel(const float* __restrict__ q, const float* __restrict__ kv1,
                                 float* __restrict__ ialpha, float* __restrict__ mblk) {
  int b = blockIdx.x * blockDim.x + threadIdx.x;
  if (b >= Bsz) return;
  float kv[64];
#pragma unroll
  for (int e = 0; e < 64; e++) kv[e] = kv1[(size_t)b * 1088 + e];
  float s[NB];
  for (int k = 0; k < NB; k++) {
    float a = 0.f;
#pragma unroll
    for (int e = 0; e < 64; e++) a += q[(size_t)b * 512 + k * 64 + e] * kv[e];
    s[k] = a * 0.125f;  // 1/sqrt(64)
  }
  for (int k = 0; k < NB; k++) {
    int rank = 0;
    for (int j = 0; j < NB; j++)
      if (s[j] < s[k] || (s[j] == s[k] && j < k)) rank++;
    mblk[(size_t)b * NB + k]   = (rank < 4) ? 0.f : 1.f;
    ialpha[(size_t)b * NB + k] = sigm(s[k]);  // iatt[:,:,1]
  }
}

// ==================== gates GEMM + fused LSTM (async B, double buffered) ====================
// tile 128 rows x (4 gates x 32 j); z = LSTM block. 40 K-tiles: 0..31 = ialpha*v1 x Wih^T,
// 32..39 = hx x Whh^T.
__global__ __launch_bounds__(256) void gates_lstm_kernel(
    const float* __restrict__ kv1, const _Float16* __restrict__ hx_h,
    const _Float16* __restrict__ wihF, const _Float16* __restrict__ whhF,
    const float* __restrict__ ialpha, const float* __restrict__ b_ih,
    const float* __restrict__ b_hh, const float* __restrict__ cx, float* __restrict__ hnew,
    float* __restrict__ cnew, _Float16* __restrict__ hnew_h) {
  __shared__ _Float16 aL[2][128 * 32];
  __shared__ _Float16 bL[2][8 * 32 * 16];
  __shared__ float    cL[128 * 132];
  const int tid = threadIdx.x, lane = tid & 31, wave = tid >> 5;
  const int wm = wave >> 1, wn = wave & 1;  // 4(M) x 2(N); wave tile 32x64
  const int m0 = blockIdx.x * 128, j0 = blockIdx.y * 32, kblk = blockIdx.z;
  const int jt8 = blockIdx.y * 8;  // ns0 within the permuted 1024-column space

  const int arow  = (tid >> 5) * 16 + (lane & 15);
  const int ahalf = lane >> 4;
  const float asc = ialpha[(size_t)(m0 + arow) * NB + kblk];
  const float*    Av = kv1 + (size_t)(m0 + arow) * 1088 + 64 + ahalf * 8;
  const _Float16* Ah = hx_h + (size_t)(m0 + arow) * NHID + kblk * BS + ahalf * 8;
  const _Float16* B1 = wihF + ((size_t)kblk * 32 * 64 + jt8) * 512 + tid * 16;
  const _Float16* B2 = whhF + ((size_t)kblk * 8 * 64 + jt8) * 512 + tid * 16;

  auto stageA = [&](int kt, int buf) {
    if (kt < 32) {  // f32 v1 row-scaled by ialpha -> f16 (VALU, overlaps async B)
      const float* sp = Av + kt * 32;
      v8h h0, h1;
#pragma unroll
      for (int e = 0; e < 8; e++) h0[e] = (_Float16)(sp[e] * asc);
#pragma unroll
      for (int e = 0; e < 8; e++) h1[e] = (_Float16)(sp[16 + e] * asc);
      *(v8h*)&aL[buf][tid * 16]     = h0;
      *(v8h*)&aL[buf][tid * 16 + 8] = h1;
    } else {
      const _Float16* sp = Ah + (kt - 32) * 32;
      async_ld_b128(&aL[buf][tid * 16], sp);
      async_ld_b128(&aL[buf][tid * 16 + 8], sp + 16);
    }
  };
  auto stageB = [&](int kt, int buf) {
    const _Float16* sp = (kt < 32) ? (B1 + (size_t)kt * 64 * 512)
                                   : (B2 + (size_t)(kt - 32) * 64 * 512);
    async_ld_b128(&bL[buf][tid * 16], sp);
    async_ld_b128(&bL[buf][tid * 16 + 8], sp + 8);
  };

  v8f acc[2][4] = {};
  stageA(0, 0);
  stageB(0, 0);
  wait_async0();
  __syncthreads();
  for (int kt = 0; kt < 40; kt++) {
    const int buf = kt & 1;
    if (kt + 1 < 40) {
      stageA(kt + 1, buf ^ 1);
      stageB(kt + 1, buf ^ 1);
    }
    v16h a0 = *(const v16h*)&aL[buf][((wm * 2 + 0) * 32 + lane) * 16];
    v16h a1 = *(const v16h*)&aL[buf][((wm * 2 + 1) * 32 + lane) * 16];
#pragma unroll
    for (int fn = 0; fn < 4; fn++) {
      v16h bf = *(const v16h*)&bL[buf][((wn * 4 + fn) * 32 + lane) * 16];
      acc[0][fn] = WMMA16(a0, bf, acc[0][fn]);
      acc[1][fn] = WMMA16(a1, bf, acc[1][fn]);
    }
    wait_async0();
    __syncthreads();
  }
  // C tile -> LDS, then fused LSTM pointwise
#pragma unroll
  for (int fm = 0; fm < 2; fm++) {
    int rr = wm * 32 + fm * 16 + (lane >> 4) * 8;
#pragma unroll
    for (int fn = 0; fn < 4; fn++) {
      int cc = wn * 64 + fn * 16 + (lane & 15);
#pragma unroll
      for (int e = 0; e < 8; e++) cL[(rr + e) * 132 + cc] = acc[fm][fn][e];
    }
  }
  __syncthreads();
  for (int idx = tid; idx < 128 * 32; idx += 256) {
    int ml = idx & 127, jl = idx >> 7;
    int j = j0 + jl;
    const size_t gb = (size_t)kblk * 1024;
    float ig = cL[ml * 132 + jl]      + b_ih[gb + j]       + b_hh[gb + j];
    float fg = cL[ml * 132 + 32 + jl] + b_ih[gb + 256 + j] + b_hh[gb + 256 + j];
    float gg = cL[ml * 132 + 64 + jl] + b_ih[gb + 512 + j] + b_hh[gb + 512 + j];
    float og = cL[ml * 132 + 96 + jl] + b_ih[gb + 768 + j] + b_hh[gb + 768 + j];
    size_t hidx = (size_t)(m0 + ml) * NHID + kblk * BS + j;
    float cold = cx[hidx];
    float cn = sigm(fg) * cold + sigm(ig) * tanhf(gg);
    float hn = sigm(og) * tanhf(cn);
    cnew[hidx]   = cn;
    hnew[hidx]   = hn;
    hnew_h[hidx] = (_Float16)hn;
  }
}

// ==================== tiny 4-head 8x8 attention (TDM-staged qkvm slab) ====================
__global__ __launch_bounds__(256) void attn_mini_kernel(const float* __restrict__ qkvm,
                                                        _Float16* __restrict__ o_h) {
  __shared__ float sm[8 * 1536];  // 48 KB slab: 8 batch rows of qkvm
  const int b0 = blockIdx.x * 8;
#if __has_builtin(__builtin_amdgcn_tensor_load_to_lds)
  if (threadIdx.x < 32) {  // wave 0 issues one TDM 1-row tile copy of 12288 f32
    const unsigned ga_lo = (unsigned)(uintptr_t)(qkvm + (size_t)b0 * 1536);
    const unsigned ga_hi = (unsigned)(((uintptr_t)(qkvm + (size_t)b0 * 1536)) >> 32);
    const unsigned la    = (unsigned)(uintptr_t)&sm[0];
    // D# group0: count=1 | lds_addr | global_addr[56:0] | type=2
    u32x4 g0 = {1u, la, ga_lo, (ga_hi & 0x01FFFFFFu) | 0x80000000u};
    // D# group1: data_size=4B; tensor_dim0=tile_dim0=12288; tensor_dim1=1; stride0=12288
    i32x8 g1 = {0x00020000, 0x30000000, 0x00010000, 0x30000000, 0, 12288, 0, 0};
    i32x4 g2 = {0, 0, 0, 0};
    i32x4 g3 = {0, 0, 0, 0};
    i32x8 g4 = {0, 0, 0, 0, 0, 0, 0, 0};  // 6-arg toolchain variant: extra group
    __builtin_amdgcn_tensor_load_to_lds(g0, g1, g2, g3, g4, 0);
    __builtin_amdgcn_s_wait_tensorcnt(0);
  }
  __syncthreads();
#else
  for (int i = threadIdx.x; i < 8 * 1536; i += 256) sm[i] = qkvm[(size_t)b0 * 1536 + i];
  __syncthreads();
#endif
  const int bi = threadIdx.x >> 5, sub = threadIdx.x & 31;
  const int h = sub >> 3, kq = sub & 7;
  const float* base = &sm[bi * 1536];
  float qv[16];
#pragma unroll
  for (int e = 0; e < 16; e++) qv[e] = base[kq * 192 + h * 16 + e];
  float lg[8], mx = -1e30f;
  for (int kk = 0; kk < 8; kk++) {
    float a = 0.f;
#pragma unroll
    for (int e = 0; e < 16; e++) a += qv[e] * base[kk * 192 + 64 + h * 16 + e];
    lg[kk] = a * 0.25f;  // 1/sqrt(16)
    mx = fmaxf(mx, lg[kk]);
  }
  float ssum = 0.f;
  for (int kk = 0; kk < 8; kk++) { lg[kk] = __expf(lg[kk] - mx); ssum += lg[kk]; }
  float inv = 1.f / ssum;
  float ov[16];
#pragma unroll
  for (int e = 0; e < 16; e++) ov[e] = 0.f;
  for (int kk = 0; kk < 8; kk++) {
    float a = lg[kk] * inv;
#pragma unroll
    for (int e = 0; e < 16; e++) ov[e] += a * base[kk * 192 + 128 + h * 16 + e];
  }
  const int b = b0 + bi;
#pragma unroll
  for (int e = 0; e < 16; e++)
    o_h[((size_t)b * NB + kq) * 64 + h * 16 + e] = (_Float16)ov[e];
}

// ==================== output projection GEMM + fused final blend ====================
// M = 32768 (b*8+k), K=64, B = Wgf fragments (gate/fc paired columns).
__global__ __launch_bounds__(256) void attout_final_kernel(
    const _Float16* __restrict__ o_h, const _Float16* __restrict__ wgfF,
    const float* __restrict__ gate_b, const float* __restrict__ fc_b,
    const float* __restrict__ hnew, const float* __restrict__ cnew,
    const float* __restrict__ mblk, const float* __restrict__ hx, const float* __restrict__ cx,
    float* __restrict__ out) {
  __shared__ _Float16 aL[2][128 * 32];
  __shared__ _Float16 bL[2][8 * 32 * 16];
  __shared__ float    cL[128 * 132];
  const int tid = threadIdx.x, lane = tid & 31, wave = tid >> 5;
  const int wm = wave >> 1, wn = wave & 1;
  const int m0 = blockIdx.x * 128, j0 = blockIdx.y * 64, ns0 = blockIdx.y * 8;

  const int arow  = (tid >> 5) * 16 + (lane & 15);
  const int ahalf = lane >> 4;
  const _Float16* Ab = o_h + (size_t)(m0 + arow) * 64 + ahalf * 8;
  const _Float16* Bb = wgfF + (size_t)ns0 * 512 + tid * 16;

  v8f acc[2][4] = {};
  async_ld_b128(&aL[0][tid * 16], Ab);
  async_ld_b128(&aL[0][tid * 16 + 8], Ab + 16);
  async_ld_b128(&bL[0][tid * 16], Bb);
  async_ld_b128(&bL[0][tid * 16 + 8], Bb + 8);
  wait_async0();
  __syncthreads();
  for (int kt = 0; kt < 2; kt++) {
    const int buf = kt & 1;
    if (kt == 0) {
      async_ld_b128(&aL[1][tid * 16], Ab + 32);
      async_ld_b128(&aL[1][tid * 16 + 8], Ab + 48);
      async_ld_b128(&bL[1][tid * 16], Bb + 32L * 512);
      async_ld_b128(&bL[1][tid * 16 + 8], Bb + 32L * 512 + 8);
    }
    v16h a0 = *(const v16h*)&aL[buf][((wm * 2 + 0) * 32 + lane) * 16];
    v16h a1 = *(const v16h*)&aL[buf][((wm * 2 + 1) * 32 + lane) * 16];
#pragma unroll
    for (int fn = 0; fn < 4; fn++) {
      v16h bf = *(const v16h*)&bL[buf][((wn * 4 + fn) * 32 + lane) * 16];
      acc[0][fn] = WMMA16(a0, bf, acc[0][fn]);
      acc[1][fn] = WMMA16(a1, bf, acc[1][fn]);
    }
    wait_async0();
    __syncthreads();
  }
#pragma unroll
  for (int fm = 0; fm < 2; fm++) {
    int rr = wm * 32 + fm * 16 + (lane >> 4) * 8;
#pragma unroll
    for (int fn = 0; fn < 4; fn++) {
      int cc = wn * 64 + fn * 16 + (lane & 15);
#pragma unroll
      for (int e = 0; e < 8; e++) cL[(rr + e) * 132 + cc] = acc[fm][fn][e];
    }
  }
  __syncthreads();
  float* hxo = out;
  float* cxo = out + HTOT;
  float* mo  = out + 2 * HTOT;
  for (int idx = tid; idx < 128 * 64; idx += 256) {
    int ml = idx & 127, tp = idx >> 7;
    int j = j0 + tp;
    float sg  = cL[ml * 132 + 2 * tp]     + gate_b[j];
    float sf  = cL[ml * 132 + 2 * tp + 1] + fc_b[j];
    float att = sigm(sg) * tanhf(sf);
    int m = m0 + ml, b = m >> 3, kblk = m & 7;
    size_t hidx = (size_t)b * NHID + kblk * BS + j;
    float mk = mblk[(size_t)b * NB + kblk];
    float hf = hnew[hidx] + att;
    hxo[hidx] = mk * hf + (1.f - mk) * hx[hidx];
    cxo[hidx] = mk * cnew[hidx] + (1.f - mk) * cx[hidx];
    mo[hidx]  = mk;
  }
}

// ==================== host orchestration ====================
extern "C" void kernel_launch(void* const* d_in, const int* in_sizes, int n_in, void* d_out,
                              int out_size, void* d_ws, size_t ws_size, hipStream_t stream) {
  const float* inp    = (const float*)d_in[0];
  const float* hx     = (const float*)d_in[1];
  const float* cx     = (const float*)d_in[2];
  const float* Wq_i   = (const float*)d_in[3];
  const float* Wk_i   = (const float*)d_in[4];
  const float* Wv_i   = (const float*)d_in[5];
  const float* Wq_m   = (const float*)d_in[6];
  const float* Wk_m   = (const float*)d_in[7];
  const float* Wv_m   = (const float*)d_in[8];
  const float* fc_w   = (const float*)d_in[9];
  const float* fc_b   = (const float*)d_in[10];
  const float* gate_w = (const float*)d_in[11];
  const float* gate_b = (const float*)d_in[12];
  const float* Wih    = (const float*)d_in[13];
  const float* Whh    = (const float*)d_in[14];
  const float* b_ih   = (const float*)d_in[15];
  const float* b_hh   = (const float*)d_in[16];
  // d_in[17] = step (unused)

  char*  ws  = (char*)d_ws;
  size_t off = 0;
  auto alloc = [&](size_t bytes) -> void* {
    void* p = ws + off;
    off += (bytes + 255) & ~(size_t)255;
    return p;
  };
  _Float16* inp_h  = (_Float16*)alloc((size_t)Bsz * NINP * 2);
  _Float16* hx_h   = (_Float16*)alloc(HTOT * 2);
  _Float16* wkv1F  = (_Float16*)alloc((size_t)NINP * 1088 * 2);
  _Float16* wqiF   = (_Float16*)alloc((size_t)NB * BS * DKI * 2);
  _Float16* wihF   = (_Float16*)alloc((size_t)NB * 1024 * 1024 * 2);
  _Float16* whhF   = (_Float16*)alloc((size_t)NB * 256 * 1024 * 2);
  _Float16* wqkvmF = (_Float16*)alloc((size_t)NB * BS * 192 * 2);
  _Float16* wgfF   = (_Float16*)alloc((size_t)64 * 512 * 2);
  float*    kv1    = (float*)alloc((size_t)Bsz * 1088 * 4);  // [:,0:64]=k1, [:,64:]=v1
  float*    qws    = (float*)alloc((size_t)Bsz * 512 * 4);
  float*    ialpha = (float*)alloc((size_t)Bsz * NB * 4);
  float*    mblk   = (float*)alloc((size_t)Bsz * NB * 4);
  float*    hnew   = (float*)alloc(HTOT * 4);
  float*    cnew   = (float*)alloc(HTOT * 4);
  _Float16* hnew_h = (_Float16*)alloc(HTOT * 2);
  float*    qkvm   = (float*)alloc((size_t)Bsz * 1536 * 4);
  _Float16* o_h    = (_Float16*)alloc((size_t)Bsz * NB * 64 * 2);
  (void)ws_size; (void)in_sizes; (void)n_in; (void)out_size;

  dim3 blk(256);
  cvt_f16_kernel<<<2048, blk, 0, stream>>>(inp_h, inp, (long)Bsz * NINP);
  cvt_f16_kernel<<<4096, blk, 0, stream>>>(hx_h, hx, (long)HTOT);
  pack_wkv1_kernel<<<1024, blk, 0, stream>>>(wkv1F, Wk_i, Wv_i);
  pack_wqi_kernel<<<256, blk, 0, stream>>>(wqiF, Wq_i);
  pack_wih_kernel<<<4096, blk, 0, stream>>>(wihF, Wih);
  pack_whh_kernel<<<2048, blk, 0, stream>>>(whhF, Whh);
  pack_wqkvm_kernel<<<512, blk, 0, stream>>>(wqkvmF, Wq_m, Wk_m, Wv_m);
  pack_wgf_kernel<<<128, blk, 0, stream>>>(wgfF, gate_w, fc_w);

  // kv1 = inp @ [Wk1 | Wv1]            (M=4096, N=1088, K=512)
  gemm_f16_nt<<<dim3(32, 17, 1), blk, 0, stream>>>(inp_h, NINP, 0, wkv1F, 68, 0, kv1, 1088, 0,
                                                   NINP);
  // q[b, k*64+e] = hx_blk(k) @ Wq_i[k] (M=4096, N=64, K=256, z=k)
  gemm_f16_nt<<<dim3(32, 1, NB), blk, 0, stream>>>(hx_h, NHID, BS, wqiF, 4,
                                                   (long)8 * 4 * 512, qws, 512, DKI, BS);
  iatt_topk_kernel<<<Bsz / 256, blk, 0, stream>>>(qws, kv1, ialpha, mblk);
  gates_lstm_kernel<<<dim3(32, 8, NB), blk, 0, stream>>>(kv1, hx_h, wihF, whhF, ialpha, b_ih,
                                                         b_hh, cx, hnew, cnew, hnew_h);
  // qm|km|vm = h_new_blk(k) @ [Wq_m|Wk_m|Wv_m][k]   (M=4096, N=192, K=256, z=k)
  gemm_f16_nt<<<dim3(32, 3, NB), blk, 0, stream>>>(hnew_h, NHID, BS, wqkvmF, 12,
                                                   (long)8 * 12 * 512, qkvm, 1536, 192, BS);
  attn_mini_kernel<<<Bsz / 8, blk, 0, stream>>>(qkvm, o_h);
  attout_final_kernel<<<dim3(256, 4, 1), blk, 0, stream>>>(o_h, wgfF, gate_b, fc_b, hnew, cnew,
                                                           mblk, hx, cx, (float*)d_out);
}
